// CNNMamba_21388937134716
// MI455X (gfx1250) — matrix operand order repaired and checked by reading, not compiled, not run on hardware
//
#include <hip/hip_runtime.h>
#include <hip/hip_bf16.h>

typedef _Float16 h16;
typedef __attribute__((ext_vector_type(8)))  h16   v8h;
typedef __attribute__((ext_vector_type(16))) h16   v16h;
typedef __attribute__((ext_vector_type(8)))  float v8f;

#define EPSF 1e-5f
#define SMEM_BYTES 245760   // 240 KB of the 320 KB WGP LDS

__device__ __forceinline__ float sigm_(float x){ return 1.f/(1.f+__expf(-x)); }
__device__ __forceinline__ float silu_(float x){ return x*sigm_(x); }
__device__ __forceinline__ float gelu_(float x){ return 0.5f*x*(1.f+erff(x*0.70710678118f)); }
__device__ __forceinline__ float softplus_(float x){ return (x>20.f)?x:log1pf(__expf(x)); }

// ---- WMMA tile loaders (CDNA5 16x16x32 f16 layouts, wave32) ----
// A (16x32, MxK) from LDS: lane l -> m=l&15, g=l>>4; halves i = K[g*16 + i]
// (0..15 contiguous). Two 16B ds_load_b128 per lane.
__device__ __forceinline__ v16h load_a_lds(const h16* A, int ldA, int m0, int k0, int lane){
  const int m = lane & 15, g = lane >> 4;
  const h16* p = A + (m0 + m) * ldA + k0 + g * 16;
  v8h lo = *(const v8h*)p;
  v8h hi = *(const v8h*)(p + 8);
  return __builtin_shufflevector(lo, hi, 0,1,2,3,4,5,6,7,8,9,10,11,12,13,14,15);
}
// B (32x16, KxN) with W stored row-major (N x K) f16 in global: lane l -> n=l&15, g=l>>4;
// halves i = W[n0+n][k0 + g*16 + i] -> one contiguous 32B read per lane (L2/WGP$-resident).
__device__ __forceinline__ v16h load_b_glb(const h16* __restrict__ W, int K, int k0, int n0, int lane){
  const int n = lane & 15, g = lane >> 4;
  return *(const v16h*)(W + (size_t)(n0 + n) * K + k0 + g * 16);
}

// Tiled GEMM over the 8 waves of the workgroup: Y(Mt*16 x Nt*16) = A(LDS f16) * W^T(global f16).
// Column-major tile ownership: each wave claims an N-column, loads its KT B-tiles into
// registers ONCE, then sweeps M-tiles with only ds_load_b128 + v_wmma in the inner loop.
// For small Nt, M is split across wave groups (ms = 8/Nt) so all waves stay busy.
// Epilogue lambda gets (m0, n0, lane, acc); C layout: lane n=l&15, g=l>>4, row = m0+g*8+j.
template <int KT, typename Epi>
__device__ __forceinline__ void gemm_t(const h16* A, int ldA, const h16* __restrict__ W,
                                       int K, int Mt, int Nt, int tid, Epi epi){
  const int wave = tid >> 5, lane = tid & 31;
  int ms = 8 / Nt; if (ms < 1) ms = 1;         // M-split factor when few N-columns
  const int cstep = 8 / ms;
  const int mt0 = (ms > 1) ? (wave % ms) : 0;
  for (int nt = wave / ms; nt < Nt; nt += cstep){
    const int n0 = nt * 16;
    v16h breg[KT];
#pragma unroll
    for (int kt = 0; kt < KT; kt++) breg[kt] = load_b_glb(W, K, kt * 32, n0, lane);
    for (int mt = mt0; mt < Mt; mt += ms){
      const int m0 = mt * 16;
      v8f acc = {0.f,0.f,0.f,0.f,0.f,0.f,0.f,0.f};
#pragma unroll
      for (int kt = 0; kt < KT; kt++){
        v16h a = load_a_lds(A, ldA, m0, kt * 32, lane);
        acc = __builtin_amdgcn_wmma_f32_16x16x32_f16(false, a, false, breg[kt],
                                                     (short)0, acc, false, false);
      }
      epi(m0, n0, lane, acc);
    }
  }
}

// ---- parameter bundles ----
struct BlkP {
  const float *ln1g,*ln1b,*convw,*convb,*dtb,*D,*ln2g,*ln2b,*fb1,*fb2,*Aneg;
  const h16 *Win,*Wx,*Wdt,*Wout,*Wf1,*Wf2;
};
struct MK {
  const float* x;
  const float *c1w,*c1b,*b1g,*b1b,*b1m,*b1v,*r1w,*r1b;
  const float *c2w,*c2b,*b2g,*b2b,*b2m,*b2v,*r2w,*r2b;
  BlkP blk[2];
  const float *postg,*postb,*hw1,*hb1,*hw2,*hb2;
  float* xs;      // ws: (1024, 250, 64) f32 residual stream (L2-resident)
  float* gate;    // ws: (1024, 256, 128) f32 (f32 so the epilogue branch cannot tail-merge)
  float* out;     // (1024, 8)
};
struct PrepP {
  const float* wsrc[12]; h16* wdst[12]; int wn[12];
  const float* alog[2];  float* aneg[2];
};

// ---- prep: convert projection weights f32->f16, A = -exp(A_log) ----
__global__ void prep_kernel(PrepP p){
  const int tid = blockIdx.x * blockDim.x + threadIdx.x;
  const int stride = gridDim.x * blockDim.x;
  for (int j = 0; j < 12; j++)
    for (int i = tid; i < p.wn[j]; i += stride) p.wdst[j][i] = (h16)p.wsrc[j][i];
  for (int j = 0; j < 2; j++)
    for (int i = tid; i < 2048; i += stride) p.aneg[j][i] = -__expf(p.alog[j][i]);
}

// ---- megakernel: one workgroup = one sample, everything LDS-resident ----
__global__ __launch_bounds__(256, 1) void mamba_mega(MK P){
  __shared__ __align__(16) unsigned char smem[SMEM_BYTES];
  const int tid = threadIdx.x;
  const int s = blockIdx.x;

  h16* tn = (h16*)(smem);             // 256x64  (LN output, f16)
  h16* up = (h16*)(smem + 32768);     // 256x128 (u_pre, then delta)
  h16* ub = (h16*)(smem + 98304);     // 256x128 (u after conv+silu)
  h16* xd = (h16*)(smem + 163840);    // 256x160 (delta_raw|b|c, then y|b|c, then gated y)
  float* xin = (float*)smem;          // P0 alias: 1000x10 f32
  float* h1p = (float*)(smem + 65536);// P0 alias: 32x500 f32

  float* xs = P.xs + (size_t)s * 16000;
  float* gate = P.gate + (size_t)s * 32768;

  // ================= P0: CNN front-end =================
  const float* xg = P.x + (size_t)s * 10000;
  for (int i = tid; i < 10000; i += 256) xin[i] = xg[i];
  __syncthreads();
  // stage 1 (conv3+BN+SiLU + res1, fused maxpool2) -> h1p (32,500)
  for (int idx = tid; idx < 32 * 500; idx += 256){
    const int oc = idx / 500, j = idx % 500;
    float mx = -1e30f;
    const float inv = P.b1g[oc] * rsqrtf(P.b1v[oc] + EPSF);
    const float sh  = P.b1b[oc] - P.b1m[oc] * inv;
    for (int pp = 0; pp < 2; pp++){
      const int p = 2*j + pp;
      float acc = P.c1b[oc], res = P.r1b[oc];
      for (int ic = 0; ic < 10; ic++){
        const float xm = (p > 0)   ? xin[(p-1)*10+ic] : 0.f;
        const float x0 = xin[p*10+ic];
        const float xp = (p < 999) ? xin[(p+1)*10+ic] : 0.f;
        const float* w = P.c1w + (oc*10+ic)*3;
        acc += w[0]*xm + w[1]*x0 + w[2]*xp;
        res += P.r1w[oc*10+ic]*x0;
      }
      mx = fmaxf(mx, silu_(acc*inv + sh) + res);
    }
    h1p[oc*500+j] = mx;
  }
  __syncthreads();
  // stage 2 -> xs global, transposed to (250, 64)
  for (int idx = tid; idx < 64 * 250; idx += 256){
    const int oc = idx / 250, j = idx % 250;
    float mx = -1e30f;
    const float inv = P.b2g[oc] * rsqrtf(P.b2v[oc] + EPSF);
    const float sh  = P.b2b[oc] - P.b2m[oc] * inv;
    for (int pp = 0; pp < 2; pp++){
      const int p = 2*j + pp;
      float acc = P.c2b[oc], res = P.r2b[oc];
      for (int ic = 0; ic < 32; ic++){
        const float xm = (p > 0)   ? h1p[ic*500+p-1] : 0.f;
        const float x0 = h1p[ic*500+p];
        const float xp = (p < 499) ? h1p[ic*500+p+1] : 0.f;
        const float* w = P.c2w + (oc*32+ic)*3;
        acc += w[0]*xm + w[1]*x0 + w[2]*xp;
        res += P.r2w[oc*32+ic]*x0;
      }
      mx = fmaxf(mx, silu_(acc*inv + sh) + res);
    }
    xs[j*64+oc] = mx;
  }
  __syncthreads();

  // ================= Mamba blocks =================
  for (int blk = 0; blk < 2; blk++){
    const BlkP& B = P.blk[blk];

    auto do_ln = [&](const float* g, const float* b){
      const int r = tid;             // 256 threads, one row each
      if (r < 250){
        float mu = 0.f;
        for (int i = 0; i < 64; i++) mu += xs[r*64+i];
        mu *= (1.f/64.f);
        float var = 0.f;
        for (int i = 0; i < 64; i++){ float d = xs[r*64+i]-mu; var += d*d; }
        var *= (1.f/64.f);
        const float inv = rsqrtf(var + EPSF);
        for (int i = 0; i < 64; i++)
          tn[r*64+i] = (h16)((xs[r*64+i]-mu)*inv*g[i] + b[i]);
      } else {
        for (int i = 0; i < 64; i++) tn[r*64+i] = (h16)0.f;
      }
    };

    // --- LN1 + in_proj (256x64 @ 64x256): u_pre -> up (LDS f16), gate -> global f32 ---
    // The branches store different types/widths (b16 LDS vs b32 global) so the compiler
    // cannot tail-merge them into FLAT stores.
    do_ln(B.ln1g, B.ln1b);
    __syncthreads();
    gemm_t<2>(tn, 64, B.Win, 64, 16, 16, tid, [&](int m0,int n0,int lane,v8f acc){
      const int n = lane & 15, g = lane >> 4;
      const int r0 = m0 + g*8;
      if (n0 < 128){
        h16* dst = up + (size_t)r0*128 + n0 + n;
#pragma unroll
        for (int j = 0; j < 8; j++) dst[j*128] = (h16)acc[j];
      } else {
        float* dstg = gate + (size_t)r0*128 + (n0 - 128) + n;
#pragma unroll
        for (int j = 0; j < 8; j++) dstg[j*128] = acc[j];
      }
    });
    __syncthreads();
    // --- depthwise conv(k=3) + SiLU: up -> ub ---
    for (int idx = tid; idx < 256*128; idx += 256){
      const int l = idx >> 7, d = idx & 127;
      float v = 0.f;
      if (l < 250){
        const float xm = (l > 0)   ? (float)up[(l-1)*128+d] : 0.f;
        const float x0 = (float)up[l*128+d];
        const float xp = (l < 249) ? (float)up[(l+1)*128+d] : 0.f;
        const float* w = B.convw + d*3;
        v = silu_(w[0]*xm + w[1]*x0 + w[2]*xp + B.convb[d]);
      }
      ub[idx] = (h16)v;
    }
    __syncthreads();
    // --- x_proj (256x128 @ 128x160) -> xd = [delta_raw | b | c] ---
    gemm_t<4>(ub, 128, B.Wx, 128, 16, 10, tid, [&](int m0,int n0,int lane,v8f acc){
      const int n = lane & 15, g = lane >> 4;
      h16* dst = xd + (size_t)(m0 + g*8)*160 + n0 + n;
#pragma unroll
      for (int j = 0; j < 8; j++) dst[j*160] = (h16)acc[j];
    });
    __syncthreads();
    // --- dt proj + softplus: xd[:, :128] @ 128x128 -> up (delta) ---
    gemm_t<4>(xd, 160, B.Wdt, 128, 16, 8, tid, [&](int m0,int n0,int lane,v8f acc){
      const int n = lane & 15, g = lane >> 4;
      const float bias = B.dtb[n0 + n];
      h16* dst = up + (size_t)(m0 + g*8)*128 + n0 + n;
#pragma unroll
      for (int j = 0; j < 8; j++) dst[j*128] = (h16)softplus_(acc[j] + bias);
    });
    __syncthreads();
    // --- selective scan: 128 threads, h[16] in regs; y into xd[:, :128] ---
    if (tid < 128){
      const int d = tid;
      float a[16], h[16];
#pragma unroll
      for (int q = 0; q < 16; q++){ a[q] = B.Aneg[d*16+q]; h[q] = 0.f; }
      for (int t = 0; t < 250; t++){
        const float dl = (float)up[t*128+d];
        const float u  = (float)ub[t*128+d];
        const float du = dl * u;
        const h16* bc = xd + t*160 + 128;
        float y = 0.f;
#pragma unroll
        for (int q = 0; q < 16; q++){
          h[q] = __expf(dl*a[q])*h[q] + du*(float)bc[q];
          y += h[q] * (float)bc[16+q];
        }
        xd[t*160 + d] = (h16)y;
      }
    }
    __syncthreads();
    // --- gating: y = (y + u*D) * silu(gate), in place in xd[:, :128] ---
    for (int idx = tid; idx < 256*128; idx += 256){
      const int l = idx >> 7, d = idx & 127;
      float v = 0.f;
      if (l < 250){
        const float y = (float)xd[l*160+d];
        const float u = (float)ub[idx];
        v = (y + u*B.D[d]) * silu_(gate[idx]);
      }
      xd[l*160 + d] = (h16)v;
    }
    __syncthreads();
    // --- out_proj (256x128 @ 128x64), residual add into xs ---
    gemm_t<4>(xd, 160, B.Wout, 128, 16, 4, tid, [&](int m0,int n0,int lane,v8f acc){
      const int n = lane & 15, g = lane >> 4;
#pragma unroll
      for (int j = 0; j < 8; j++){
        const int r = m0 + g*8 + j;
        if (r < 250) xs[r*64 + n0 + n] += acc[j];
      }
    });
    __syncthreads();
    // --- LN2 + FFN: gelu(tn@W1+b1) -> up; xs += up@W2+b2 ---
    do_ln(B.ln2g, B.ln2b);
    __syncthreads();
    gemm_t<2>(tn, 64, B.Wf1, 64, 16, 8, tid, [&](int m0,int n0,int lane,v8f acc){
      const int n = lane & 15, g = lane >> 4;
      const float bias = B.fb1[n0 + n];
      h16* dst = up + (size_t)(m0 + g*8)*128 + n0 + n;
#pragma unroll
      for (int j = 0; j < 8; j++) dst[j*128] = (h16)gelu_(acc[j] + bias);
    });
    __syncthreads();
    gemm_t<4>(up, 128, B.Wf2, 128, 16, 4, tid, [&](int m0,int n0,int lane,v8f acc){
      const int n = lane & 15, g = lane >> 4;
      const float bias = B.fb2[n0 + n];
#pragma unroll
      for (int j = 0; j < 8; j++){
        const int r = m0 + g*8 + j;
        if (r < 250) xs[r*64 + n0 + n] += acc[j] + bias;
      }
    });
    __syncthreads();
  }

  // ================= P3: post-LN, mean over L, head =================
  float* accum = (float*)smem;          // 64
  float* vecb  = (float*)(smem + 256);  // 64
  float* hid   = (float*)(smem + 512);  // 64
  for (int i = tid; i < 64; i += 256) accum[i] = 0.f;
  __syncthreads();
  if (tid < 250){
    const int r = tid;
    float mu = 0.f;
    for (int i = 0; i < 64; i++) mu += xs[r*64+i];
    mu *= (1.f/64.f);
    float var = 0.f;
    for (int i = 0; i < 64; i++){ float d = xs[r*64+i]-mu; var += d*d; }
    var *= (1.f/64.f);
    const float inv = rsqrtf(var + EPSF);
    for (int i = 0; i < 64; i++){
      const float v = (xs[r*64+i]-mu)*inv*P.postg[i] + P.postb[i];
      atomicAdd(&accum[i], v);          // ds_add_f32
    }
  }
  __syncthreads();
  if (tid < 64) vecb[tid] = accum[tid] * (1.f/250.f);
  __syncthreads();
  if (tid < 64){
    float acc = P.hb1[tid];
    for (int j = 0; j < 64; j++) acc += vecb[j]*P.hw1[tid*64+j];
    hid[tid] = gelu_(acc);
  }
  __syncthreads();
  if (tid < 8){
    float acc = P.hb2[tid];
    for (int j = 0; j < 64; j++) acc += hid[j]*P.hw2[tid*64+j];
    P.out[s*8 + tid] = acc;
  }
}

extern "C" void kernel_launch(void* const* d_in, const int* in_sizes, int n_in,
                              void* d_out, int out_size, void* d_ws, size_t ws_size,
                              hipStream_t stream){
  (void)in_sizes; (void)n_in; (void)out_size; (void)ws_size;
  auto F = [&](int i){ return (const float*)d_in[i]; };
  unsigned char* ws = (unsigned char*)d_ws;

  // workspace carving (256B aligned)
  size_t off = 0;
  auto carve = [&](size_t bytes){ size_t o = off; off += (bytes + 255) & ~(size_t)255; return o; };
  const int wN[6] = {256*64, 160*128, 128*128, 64*128, 128*64, 64*128};
  size_t wOff[2][6];
  for (int b = 0; b < 2; b++) for (int j = 0; j < 6; j++) wOff[b][j] = carve((size_t)wN[j]*2);
  size_t aOff[2]; for (int b = 0; b < 2; b++) aOff[b] = carve(2048*4);
  size_t xsOff = carve((size_t)1024*16000*4);
  size_t gOff  = carve((size_t)1024*32768*4);   // gate f32

  // input index map (setup_inputs dict insertion order, blocks at 17 and 34)
  const int wIdxInBlk[6] = {2, 5, 6, 8, 13, 15}; // in_proj, x_proj, dt, out_proj, ffn1, ffn2

  PrepP pp{};
  for (int b = 0; b < 2; b++){
    const int base = 17 + b*17;
    for (int j = 0; j < 6; j++){
      pp.wsrc[b*6+j] = F(base + wIdxInBlk[j]);
      pp.wdst[b*6+j] = (h16*)(ws + wOff[b][j]);
      pp.wn[b*6+j]   = wN[j];
    }
    pp.alog[b] = F(base + 9);
    pp.aneg[b] = (float*)(ws + aOff[b]);
  }
  prep_kernel<<<64, 256, 0, stream>>>(pp);

  MK mk{};
  mk.x = F(0);
  mk.c1w=F(1);  mk.c1b=F(2);  mk.b1g=F(3);  mk.b1b=F(4);  mk.b1m=F(5);  mk.b1v=F(6);  mk.r1w=F(7);  mk.r1b=F(8);
  mk.c2w=F(9);  mk.c2b=F(10); mk.b2g=F(11); mk.b2b=F(12); mk.b2m=F(13); mk.b2v=F(14); mk.r2w=F(15); mk.r2b=F(16);
  for (int b = 0; b < 2; b++){
    const int base = 17 + b*17;
    mk.blk[b].ln1g = F(base+0);  mk.blk[b].ln1b = F(base+1);
    mk.blk[b].convw= F(base+3);  mk.blk[b].convb= F(base+4);
    mk.blk[b].dtb  = F(base+7);
    mk.blk[b].D    = F(base+10);
    mk.blk[b].ln2g = F(base+11); mk.blk[b].ln2b = F(base+12);
    mk.blk[b].fb1  = F(base+14); mk.blk[b].fb2  = F(base+16);
    mk.blk[b].Aneg = (const float*)(ws + aOff[b]);
    mk.blk[b].Win  = (const h16*)(ws + wOff[b][0]);
    mk.blk[b].Wx   = (const h16*)(ws + wOff[b][1]);
    mk.blk[b].Wdt  = (const h16*)(ws + wOff[b][2]);
    mk.blk[b].Wout = (const h16*)(ws + wOff[b][3]);
    mk.blk[b].Wf1  = (const h16*)(ws + wOff[b][4]);
    mk.blk[b].Wf2  = (const h16*)(ws + wOff[b][5]);
  }
  mk.postg=F(51); mk.postb=F(52); mk.hw1=F(53); mk.hb1=F(54); mk.hw2=F(55); mk.hb2=F(56);
  mk.xs   = (float*)(ws + xsOff);
  mk.gate = (float*)(ws + gOff);
  mk.out  = (float*)d_out;
  mamba_mega<<<1024, 256, 0, stream>>>(mk);
}
